// ExternalAttention_59425167507704
// MI455X (gfx1250) — matrix operand order repaired and checked by reading, not compiled, not run on hardware
//
#include <hip/hip_runtime.h>

#define D_MODELc 1024
#define N_HEADSc 16
#define D_HEADc  64
#define MEM_Sc   256
#define BATCHc   4
#define SEQ_Tc   4096
#define BTc      (BATCHc * SEQ_Tc)   // 16384 rows

typedef __attribute__((ext_vector_type(16))) __bf16 v16bf;
typedef __attribute__((ext_vector_type(8)))  __bf16 v8bf;
typedef __attribute__((ext_vector_type(8)))  float  v8f;
typedef __attribute__((ext_vector_type(4)))  float  f4;
typedef __attribute__((ext_vector_type(4)))  unsigned int u4;

__device__ __forceinline__ unsigned short f2bf(float f) {
  unsigned u = __float_as_uint(f);
  u += 0x7fffu + ((u >> 16) & 1u);          // round-to-nearest-even
  return (unsigned short)(u >> 16);
}
__device__ __forceinline__ unsigned pk2(float a, float b) {
  return (unsigned)f2bf(a) | ((unsigned)f2bf(b) << 16);
}
__device__ __forceinline__ v8f zero8() {
  v8f z; for (int i = 0; i < 8; ++i) z[i] = 0.f; return z;
}

union FragU { v16bf v; u4 q[2]; };

// A-matrix 16xK(bf16) fragment: lane m=lane&15; half h -> k = 16*(h>>3) + 8*(lane>>4) + (h&7)
__device__ __forceinline__ v16bf load_a_frag(const unsigned short* lds, int ld, int row0, int kbase) {
  int lane = threadIdx.x & 31;
  const unsigned short* p = lds + (row0 + (lane & 15)) * ld + kbase + ((lane >> 4) << 3);
  FragU f;
  f.q[0] = *(const u4*)(p);        // h=0..7  : k = off+0..7
  f.q[1] = *(const u4*)(p + 16);   // h=8..15 : k = 16+off+0..7
  return f.v;
}
// B-matrix Kx16(bf16) fragment, stored as Bt[n][k]: lane n=lane&15; k = 16*(h>>3) + 8*(lane>>4) + (h&7)
__device__ __forceinline__ v16bf load_b_frag(const unsigned short* lds, int ld, int row0, int kbase) {
  int lane = threadIdx.x & 31;
  const unsigned short* p = lds + (row0 + (lane & 15)) * ld + kbase + ((lane >> 4) << 3);
  FragU f;
  f.q[0] = *(const u4*)(p);        // h=0..7
  f.q[1] = *(const u4*)(p + 16);   // h=8..15
  return f.v;
}

#define WMMA_BF16(a, b, c) \
  __builtin_amdgcn_wmma_f32_16x16x32_bf16(false, (a), false, (b), (short)0, (c), false, false)

// 16B-per-lane async DMA: global -> LDS, tracked by ASYNCcnt (CDNA5 TDM-lite path)
__device__ __forceinline__ void async_ld_b128(const unsigned short* gptr, const unsigned short* lptr) {
  unsigned long long ga = (unsigned long long)gptr;
  unsigned la = (unsigned)(unsigned long long)(const void*)lptr;   // low 32b = LDS offset
  asm volatile("global_load_async_to_lds_b128 %0, %1, off" :: "v"(la), "v"(ga) : "memory");
}
__device__ __forceinline__ void wait_async0() {
  asm volatile("s_wait_asynccnt 0x0" ::: "memory");
}

// ---------------------------------------------------------------------------
// f32 -> bf16 bulk convert (one-time; n must be a multiple of 8)
// ---------------------------------------------------------------------------
__global__ __launch_bounds__(256) void f32_to_bf16_kernel(
    const float* __restrict__ src, unsigned short* __restrict__ dst, int n)
{
  int i = (blockIdx.x * 256 + threadIdx.x) * 8;
  if (i >= n) return;
  f4 a = *(const f4*)(src + i);
  f4 b = *(const f4*)(src + i + 4);
  u4 u = { pk2(a[0], a[1]), pk2(a[2], a[3]), pk2(b[0], b[1]), pk2(b[2], b[3]) };
  *(u4*)(dst + i) = u;
}

// ---------------------------------------------------------------------------
// C[M,N](f32|bf16) = A[M,K](bf16) * Bw[N,K](bf16)^T
// Block tile 128x128, k-step 32, double-buffered LDS filled by async DMA,
// 8 waves (2x4), 64x32 per wave. Staging overlaps WMMA; one
// s_wait_asynccnt(0) + barrier per k-step.
// ---------------------------------------------------------------------------
template <bool OUT_BF16>
__global__ __launch_bounds__(256) void gemm_bt_async_kernel(
    const unsigned short* __restrict__ Abf, const unsigned short* __restrict__ Bbf,
    void* __restrict__ Cp, int M, int N, int K)
{
  constexpr int BM = 128, BN = 128, BK = 32, LD = 40;  // LD pad keeps 16B alignment
  __shared__ unsigned short As[2][BM * LD];
  __shared__ unsigned short Bs[2][BN * LD];

  const int tid  = threadIdx.x;
  const int lane = tid & 31;
  const int wid  = tid >> 5;
  const int wm   = (wid & 1) * 64;
  const int wn   = (wid >> 1) * 32;
  const int m0   = blockIdx.y * BM;
  const int n0   = blockIdx.x * BN;

  // each thread DMAs 2 x 16B per matrix per k-step (tile = 128 rows x 32 bf16)
  const int crow = tid >> 2;             // 0..63
  const int ccol = (tid & 3) * 8;        // 0,8,16,24

  auto stage_async = [&](int buf, int k0) {
    const unsigned short* ga0 = Abf + (size_t)(m0 + crow) * K + k0 + ccol;
    const unsigned short* ga1 = Abf + (size_t)(m0 + 64 + crow) * K + k0 + ccol;
    const unsigned short* gb0 = Bbf + (size_t)(n0 + crow) * K + k0 + ccol;
    const unsigned short* gb1 = Bbf + (size_t)(n0 + 64 + crow) * K + k0 + ccol;
    async_ld_b128(ga0, &As[buf][crow * LD + ccol]);
    async_ld_b128(ga1, &As[buf][(64 + crow) * LD + ccol]);
    async_ld_b128(gb0, &Bs[buf][crow * LD + ccol]);
    async_ld_b128(gb1, &Bs[buf][(64 + crow) * LD + ccol]);
  };

  v8f acc[4][2];
  for (int i = 0; i < 4; ++i) for (int j = 0; j < 2; ++j) acc[i][j] = zero8();

  const int NK = K / BK;
  stage_async(0, 0);
  wait_async0();
  __syncthreads();

  for (int ks = 0; ks < NK; ++ks) {
    const bool more = (ks + 1) < NK;
    if (more) stage_async((ks + 1) & 1, (ks + 1) * BK);  // DMA next tile during compute

    const unsigned short* la = As[ks & 1];
    const unsigned short* lb = Bs[ks & 1];
    v16bf afr[4];
    for (int mi = 0; mi < 4; ++mi) afr[mi] = load_a_frag(la, LD, wm + mi * 16, 0);
    for (int ni = 0; ni < 2; ++ni) {
      v16bf bfr = load_b_frag(lb, LD, wn + ni * 16, 0);
      for (int mi = 0; mi < 4; ++mi) acc[mi][ni] = WMMA_BF16(afr[mi], bfr, acc[mi][ni]);
    }

    if (more) wait_async0();   // our DMA for buf (ks+1)&1 is done before the barrier
    __syncthreads();
  }

  const int mb = m0 + wm + ((lane >> 4) << 3);
  const int nb = n0 + wn + (lane & 15);
  for (int mi = 0; mi < 4; ++mi)
    for (int ni = 0; ni < 2; ++ni)
      for (int r = 0; r < 8; ++r) {
        size_t idx = (size_t)(mb + mi * 16 + r) * N + (nb + ni * 16);
        if (OUT_BF16) ((unsigned short*)Cp)[idx] = f2bf(acc[mi][ni][r]);
        else          ((float*)Cp)[idx] = acc[mi][ni][r];
      }
}

// ---------------------------------------------------------------------------
// Pass 1: logits = Q @ Mk^T, softmax over S, deterministic per-block colsums
// block = 256 thr (8 waves), 128 tokens; wave handles 16 token rows x 256 s
// ---------------------------------------------------------------------------
__global__ __launch_bounds__(256) void attn_colsum_kernel(
    const unsigned short* __restrict__ Qg, const float* __restrict__ Mk,
    float* __restrict__ cpart)
{
  constexpr int LQ = 72, LK = 72;
  __shared__ unsigned short Qs[128 * LQ];
  __shared__ unsigned short Ks[MEM_Sc * LK];
  __shared__ float part[8][MEM_Sc];

  const int tb = blockIdx.x, h = blockIdx.y, b = blockIdx.z;
  const int bh = b * N_HEADSc + h;
  const int tid = threadIdx.x, lane = tid & 31, w = tid >> 5;

  {  // stage Q tile: 128 rows x 64 bf16
    const int row = tid >> 1, hf = (tid & 1) * 32;
    const unsigned short* g =
        Qg + ((size_t)(b * SEQ_Tc + tb * 128 + row)) * D_MODELc + h * D_HEADc + hf;
    unsigned short* d = &Qs[row * LQ + hf];
    *(u4*)d        = *(const u4*)g;        *(u4*)(d + 8)  = *(const u4*)(g + 8);
    *(u4*)(d + 16) = *(const u4*)(g + 16); *(u4*)(d + 24) = *(const u4*)(g + 24);
  }
  {  // stage Mk (f32 -> bf16): 256 rows x 64
    const float* g = Mk + ((size_t)h * MEM_Sc + tid) * D_HEADc;
    unsigned short* d = &Ks[tid * LK];
    for (int c = 0; c < 64; c += 8) {
      f4 a = *(const f4*)(g + c);
      f4 b2 = *(const f4*)(g + c + 4);
      u4 u = { pk2(a[0], a[1]), pk2(a[2], a[3]), pk2(b2[0], b2[1]), pk2(b2[2], b2[3]) };
      *(u4*)(d + c) = u;
    }
  }
  __syncthreads();

  v8f acc[16];
  for (int j = 0; j < 16; ++j) acc[j] = zero8();
  for (int ks = 0; ks < 2; ++ks) {
    v16bf afr = load_a_frag(Qs, LQ, w * 16, ks * 32);
    for (int j = 0; j < 16; ++j) {
      v16bf bfr = load_b_frag(Ks, LK, j * 16, ks * 32);
      acc[j] = WMMA_BF16(afr, bfr, acc[j]);
    }
  }

  const float scale = 0.125f;  // 1/sqrt(64)
  float rmax[8], rinv[8];
  for (int r = 0; r < 8; ++r) {
    float m = -3.4e38f;
    for (int j = 0; j < 16; ++j) m = fmaxf(m, acc[j][r]);
    m = fmaxf(m, __shfl_xor(m, 1)); m = fmaxf(m, __shfl_xor(m, 2));
    m = fmaxf(m, __shfl_xor(m, 4)); m = fmaxf(m, __shfl_xor(m, 8));
    rmax[r] = m;
  }
  for (int r = 0; r < 8; ++r) {
    float s = 0.f;
    for (int j = 0; j < 16; ++j) {
      float e = __expf((acc[j][r] - rmax[r]) * scale);
      acc[j][r] = e; s += e;
    }
    s += __shfl_xor(s, 1); s += __shfl_xor(s, 2);
    s += __shfl_xor(s, 4); s += __shfl_xor(s, 8);
    rinv[r] = 1.f / s;
  }
  for (int j = 0; j < 16; ++j) {  // column sums over this wave's 16 token rows
    float cs = 0.f;
    for (int r = 0; r < 8; ++r) cs += acc[j][r] * rinv[r];
    cs += __shfl_xor(cs, 16);
    if (lane < 16) part[w][j * 16 + lane] = cs;
  }
  __syncthreads();
  {  // fixed-order reduce across the 8 waves
    float s = 0.f;
    for (int w2 = 0; w2 < 8; ++w2) s += part[w2][tid];
    cpart[((size_t)bh * 32 + tb) * MEM_Sc + tid] = s;
  }
}

__global__ void colsum_finish_kernel(const float* __restrict__ cpart,
                                     float* __restrict__ invc)
{
  int idx = blockIdx.x * 256 + threadIdx.x;  // 16384 = B*H*S
  int bh = idx >> 8, s = idx & 255;
  float t = 0.f;
  for (int tb = 0; tb < 32; ++tb) t += cpart[((size_t)bh * 32 + tb) * MEM_Sc + s];
  invc[idx] = 1.f / (t + 1e-6f);
}

// ---------------------------------------------------------------------------
// Pass 2: recompute softmax, apply 1/colsum, P @ Mv -> ctx (bf16)
// block = 128 thr (4 waves), 64 tokens; LDS phase1 {Q,Mk} reused as phase2 {P,MvT}
// ---------------------------------------------------------------------------
__global__ __launch_bounds__(128) void attn_out_kernel(
    const unsigned short* __restrict__ Qg, const float* __restrict__ Mk,
    const float* __restrict__ Mv, const float* __restrict__ invc,
    unsigned short* __restrict__ ctx)
{
  constexpr int LQ = 72, LK = 72;
  __shared__ __align__(16) unsigned char smem[65536];
  unsigned short* Qs  = (unsigned short*)smem;             // 64*72*2  = 9216B
  unsigned short* Ks  = (unsigned short*)(smem + 9216);    // 256*72*2 = 36864B
  unsigned short* Pls = (unsigned short*)smem;             // phase2: 64*256*2 = 32768B
  unsigned short* MvT = (unsigned short*)(smem + 32768);   // phase2: 64*256*2 = 32768B

  const int tb = blockIdx.x, h = blockIdx.y, b = blockIdx.z;
  const int bh = b * N_HEADSc + h;
  const int tid = threadIdx.x, lane = tid & 31, w = tid >> 5;  // w in 0..3
  const int t0 = tb * 64;

  {  // stage Q tile: 64 rows x 64 bf16
    const int row = tid >> 1, hf = (tid & 1) * 32;
    const unsigned short* g =
        Qg + ((size_t)(b * SEQ_Tc + t0 + row)) * D_MODELc + h * D_HEADc + hf;
    unsigned short* d = &Qs[row * LQ + hf];
    *(u4*)d        = *(const u4*)g;        *(u4*)(d + 8)  = *(const u4*)(g + 8);
    *(u4*)(d + 16) = *(const u4*)(g + 16); *(u4*)(d + 24) = *(const u4*)(g + 24);
  }
  for (int rr = 0; rr < 2; ++rr) {  // stage Mk: 2 rows per thread
    int row = tid * 2 + rr;
    const float* g = Mk + ((size_t)h * MEM_Sc + row) * D_HEADc;
    unsigned short* d = &Ks[row * LK];
    for (int c = 0; c < 64; c += 8) {
      f4 a = *(const f4*)(g + c);
      f4 b2 = *(const f4*)(g + c + 4);
      u4 u = { pk2(a[0], a[1]), pk2(a[2], a[3]), pk2(b2[0], b2[1]), pk2(b2[2], b2[3]) };
      *(u4*)(d + c) = u;
    }
  }
  __syncthreads();

  v8f acc[16];
  for (int j = 0; j < 16; ++j) acc[j] = zero8();
  for (int ks = 0; ks < 2; ++ks) {
    v16bf afr = load_a_frag(Qs, LQ, w * 16, ks * 32);
    for (int j = 0; j < 16; ++j) {
      v16bf bfr = load_b_frag(Ks, LK, j * 16, ks * 32);
      acc[j] = WMMA_BF16(afr, bfr, acc[j]);
    }
  }

  float invj[16];
  for (int j = 0; j < 16; ++j) invj[j] = invc[bh * MEM_Sc + j * 16 + (lane & 15)];

  const float scale = 0.125f;
  float rmax[8], rinv[8];
  for (int r = 0; r < 8; ++r) {
    float m = -3.4e38f;
    for (int j = 0; j < 16; ++j) m = fmaxf(m, acc[j][r]);
    m = fmaxf(m, __shfl_xor(m, 1)); m = fmaxf(m, __shfl_xor(m, 2));
    m = fmaxf(m, __shfl_xor(m, 4)); m = fmaxf(m, __shfl_xor(m, 8));
    rmax[r] = m;
  }
  for (int r = 0; r < 8; ++r) {
    float s = 0.f;
    for (int j = 0; j < 16; ++j) {
      float e = __expf((acc[j][r] - rmax[r]) * scale);
      acc[j][r] = e; s += e;
    }
    s += __shfl_xor(s, 1); s += __shfl_xor(s, 2);
    s += __shfl_xor(s, 4); s += __shfl_xor(s, 8);
    rinv[r] = 1.f / s;
  }

  __syncthreads();  // everyone done reading Qs/Ks before aliasing as Pls/MvT

  const int mhi = (lane >> 4) << 3;
  for (int j = 0; j < 16; ++j) {  // scatter normalized P (bf16) into LDS
    int col = j * 16 + (lane & 15);
    for (int r = 0; r < 8; ++r) {
      float p = acc[j][r] * rinv[r] * invj[j];
      Pls[(w * 16 + mhi + r) * MEM_Sc + col] = f2bf(p);
    }
  }
  {  // stage MvT[d][s] = Mv[h][s][d] (bf16)
    const float* g = Mv + (size_t)h * MEM_Sc * D_HEADc;
    for (int i = 0; i < 128; ++i) {
      int e = tid * 128 + i;
      int s = e >> 6, dd = e & 63;
      MvT[dd * MEM_Sc + s] = f2bf(g[e]);
    }
  }
  __syncthreads();

  v8f acc2[4];
  for (int ni = 0; ni < 4; ++ni) acc2[ni] = zero8();
  for (int ks = 0; ks < 8; ++ks) {  // K = 256 over s
    v16bf afr = load_a_frag(Pls, MEM_Sc, w * 16, ks * 32);
    for (int ni = 0; ni < 4; ++ni) {
      v16bf bfr = load_b_frag(MvT, MEM_Sc, ni * 16, ks * 32);
      acc2[ni] = WMMA_BF16(afr, bfr, acc2[ni]);
    }
  }
  for (int ni = 0; ni < 4; ++ni)
    for (int r = 0; r < 8; ++r) {
      int t = t0 + w * 16 + mhi + r;
      size_t idx = ((size_t)(b * SEQ_Tc + t)) * D_MODELc + h * D_HEADc + ni * 16 + (lane & 15);
      ctx[idx] = f2bf(acc2[ni][r]);
    }
}

// ---------------------------------------------------------------------------
extern "C" void kernel_launch(void* const* d_in, const int* in_sizes, int n_in,
                              void* d_out, int out_size, void* d_ws, size_t ws_size,
                              hipStream_t stream)
{
  (void)in_sizes; (void)n_in; (void)out_size;
  const float* x  = (const float*)d_in[0];
  const float* Wq = (const float*)d_in[1];
  const float* Wo = (const float*)d_in[2];
  const float* Mk = (const float*)d_in[3];
  const float* Mv = (const float*)d_in[4];

  unsigned char* ws = (unsigned char*)d_ws;
  const size_t OFF_Q   = 0;            // Q bf16:   16384*1024*2 = 32 MiB
  const size_t OFF_CTX = 33554432;     // ctx bf16: 32 MiB
  const size_t OFF_CP  = 67108864;     // colsum partials: 64*32*256 f32 = 2 MiB
  const size_t OFF_INV = 69206016;     // inv colsum: 16384 f32 (64 KiB)
  const size_t OFF_XB  = 69271552;     // x bf16:   32 MiB
  const size_t OFF_WQB = 102825984;    // Wq bf16:  2 MiB
  const size_t OFF_WOB = 104923136;    // Wo bf16:  2 MiB
  const size_t WS_NEED = 107020288;
  if (ws_size < WS_NEED) return;

  unsigned short* Qws = (unsigned short*)(ws + OFF_Q);
  unsigned short* Ctx = (unsigned short*)(ws + OFF_CTX);
  float* Cpart = (float*)(ws + OFF_CP);
  float* Invc  = (float*)(ws + OFF_INV);
  unsigned short* Xb  = (unsigned short*)(ws + OFF_XB);
  unsigned short* Wqb = (unsigned short*)(ws + OFF_WQB);
  unsigned short* Wob = (unsigned short*)(ws + OFF_WOB);

  // 0) one-time bf16 conversions (bandwidth-trivial; halves GEMM read traffic)
  f32_to_bf16_kernel<<<dim3(BTc * D_MODELc / 2048), 256, 0, stream>>>(x, Xb, BTc * D_MODELc);
  f32_to_bf16_kernel<<<dim3(D_MODELc * D_MODELc / 2048), 256, 0, stream>>>(Wq, Wqb, D_MODELc * D_MODELc);
  f32_to_bf16_kernel<<<dim3(D_MODELc * D_MODELc / 2048), 256, 0, stream>>>(Wo, Wob, D_MODELc * D_MODELc);

  // 1) Q = x @ Wq^T  (bf16 out), async-DMA staged tiles
  gemm_bt_async_kernel<true>
      <<<dim3(D_MODELc / 128, BTc / 128), 256, 0, stream>>>(Xb, Wqb, Qws, BTc, D_MODELc, D_MODELc);
  // 2) softmax over S + per-128-token-block column sums (deterministic)
  attn_colsum_kernel
      <<<dim3(SEQ_Tc / 128, N_HEADSc, BATCHc), 256, 0, stream>>>(Qws, Mk, Cpart);
  // 3) finish colsum -> 1/(sum+1e-6)
  colsum_finish_kernel<<<dim3(64), 256, 0, stream>>>(Cpart, Invc);
  // 4) recompute softmax, double-normalize, P @ Mv -> ctx (bf16)
  attn_out_kernel
      <<<dim3(SEQ_Tc / 64, N_HEADSc, BATCHc), 128, 0, stream>>>(Qws, Mk, Mv, Invc, Ctx);
  // 5) out = ctx @ Wo^T (f32), async-DMA staged tiles
  gemm_bt_async_kernel<false>
      <<<dim3(D_MODELc / 128, BTc / 128), 256, 0, stream>>>(Ctx, Wob, d_out, BTc, D_MODELc, D_MODELc);
}
